// RNN_72533407694864
// MI455X (gfx1250) — compile-verified, matrix operand order checked by reading, not run on hardware
//
#include <hip/hip_runtime.h>
#include <hip/hip_bf16.h>

// ---------------------------------------------------------------------------
// Leaky ReLU-RNN on MI455X (gfx1250, wave32, WMMA), persistent-kernel version.
//
// state_{t+1} = (1-a)*state_t + a*( relu(state_t) @ W_hh + drive_t )
// drive_t     = inputs_t @ W_ih + bias + sigma*noise_t
//
// Per step: 128x1024 @ 1024x1024 GEMM = 8(M) x 64(N) tiles of 16x16, K=1024
// -> 32 chained v_wmma_f32_16x16x32_bf16 per tile. W_hh is pre-swizzled to
// bf16 WMMA-B fragment order once (2 MB, L2-resident for all 512 steps).
//
// The 512 steps are strictly serial, so instead of 512 graph-node launches
// (launch overhead ~ per-step compute), ONE persistent kernel of 64 blocks
// (co-residency guaranteed: 1 block = 8 waves + 32 KB LDS per WGP) iterates
// all steps with a device-scope atomic grid barrier between them.
// ---------------------------------------------------------------------------

typedef __bf16 bf16;
typedef __attribute__((ext_vector_type(16))) __bf16 v16bf;
typedef __attribute__((ext_vector_type(8)))  float  v8f;

constexpr int   T_STEPS = 512;
constexpr int   BATCH   = 128;
constexpr int   NINP    = 2;
constexpr int   HID     = 1024;
constexpr float ALPHA   = 0.2f;
constexpr float OMA     = 1.0f - ALPHA;
constexpr float SIGMA_REC = 0.15811388300841897f;   // sqrt(2/0.2)*0.05

constexpr int KT = HID / 32;    // 32 k-tiles of depth 32
constexpr int NT = HID / 16;    // 64 n-tiles
constexpr int MT = BATCH / 16;  // 8  m-tiles
constexpr int NBLOCKS = MT * 8; // 64 persistent blocks

// ---------------------------------------------------------------------------
// One-time: swizzle fp32 W_hh[k][n] (row-major) into bf16 WMMA B-fragments.
// Fragment order: frag[nt][kt][lane][16 bf16]; lane = n%16 + 16*halfK,
// vgpr v holds K = kt*32 + (v>>2)*16 + half*8 + (v&3)*2 and K+1 packed lo/hi.
// ---------------------------------------------------------------------------
__global__ __launch_bounds__(256) void swizzle_whh(const float* __restrict__ W,
                                                   bf16* __restrict__ out) {
    int gid  = blockIdx.x * 256 + threadIdx.x;   // 64*32*32 = 65536 fragments
    int lane = gid & 31;
    int kt   = (gid >> 5) & 31;
    int nt   = gid >> 10;
    int l    = lane & 15;
    int hf   = lane >> 4;
    int n    = nt * 16 + l;
    bf16* dst = out + (size_t)gid * 16;
#pragma unroll
    for (int v = 0; v < 8; ++v) {
        int k = kt * 32 + (v >> 2) * 16 + hf * 8 + (v & 3) * 2;
        dst[2 * v]     = (bf16)W[(size_t)k       * HID + n];
        dst[2 * v + 1] = (bf16)W[(size_t)(k + 1) * HID + n];
    }
}

// ---------------------------------------------------------------------------
// Persistent RNN kernel. Grid: 64 blocks x 256 threads (8 waves).
//   m_tile = blockIdx.x >> 3           (each block owns one 16-row M tile)
//   n_tile = (blockIdx.x & 7)*8 + wave (one 16-col N tile per wave)
// ---------------------------------------------------------------------------
__global__ __launch_bounds__(256) void rnn_persistent(
    const float* __restrict__ inputs,   // [T][B][2]
    float*       __restrict__ out,      // [T+1][B][H]; slab t = state_t
    const bf16*  __restrict__ Wb,       // swizzled bf16 W_hh fragments
    const float* __restrict__ noise,    // [T][B][H]
    const float* __restrict__ wih,      // [2][H]
    const float* __restrict__ bias,     // [H]
    unsigned*    __restrict__ bar_cnt)  // zeroed before launch; monotonic
{
    __shared__ v16bf As[KT * 32];       // 32 kt x 32 lanes x 32B = 32 KB

    const int tid  = threadIdx.x;
    const int lane = tid & 31;
    const int wave = tid >> 5;
    const int mt   = blockIdx.x >> 3;
    const int nt   = (blockIdx.x & 7) * 8 + wave;
    const int m0   = mt * 16;
    const int l    = lane & 15;
    const int hf   = lane >> 4;

    // Per-column constants for the fused drive epilogue (invariant over t).
    const int   n  = nt * 16 + l;
    const float w0 = wih[n];
    const float w1 = wih[HID + n];
    const float bn = bias[n];

    for (int t = 0; t < T_STEPS; ++t) {
        const float* prev = out   + (size_t)t       * BATCH * HID;
        float*       next = out   + (size_t)(t + 1) * BATCH * HID;
        const float* xin  = inputs + (size_t)t * BATCH * NINP;
        const float* nz   = noise  + (size_t)t * BATCH * HID;

        // ---- Stage relu(state_t) A-fragments for this M tile into LDS. ---
        {
            const float* srow = prev + (size_t)(m0 + l) * HID;  // lane = row M
#pragma unroll
            for (int j = 0; j < 4; ++j) {
                int kt = wave + j * 8;                          // 8 waves x 4
                bf16* dst = (bf16*)&As[kt * 32 + lane];
#pragma unroll
                for (int v = 0; v < 8; ++v) {
                    int k = kt * 32 + (v >> 2) * 16 + hf * 8 + (v & 3) * 2;
                    float2 s = *(const float2*)(srow + k);
                    dst[2 * v]     = (bf16)fmaxf(s.x, 0.0f);
                    dst[2 * v + 1] = (bf16)fmaxf(s.y, 0.0f);
                }
            }
        }
        __syncthreads();

        // ---- K sweep: 32 chained WMMAs; A from LDS, B from L2 Wb. --------
        v8f c = {};
        const v16bf* Af = &As[lane];
        const v16bf* Bf = ((const v16bf*)Wb) + (size_t)nt * KT * 32 + lane;
#pragma unroll 4
        for (int kt = 0; kt < KT; ++kt) {
            v16bf a = Af[kt * 32];
            v16bf b = Bf[kt * 32];
            c = __builtin_amdgcn_wmma_f32_16x16x32_bf16(
                    /*neg_a=*/false, a, /*neg_b=*/false, b,
                    /*c_mod=*/(short)0, c, /*reuse_a=*/false, /*reuse_b=*/false);
        }

        // ---- Fused epilogue: drive + leak update -> slab t+1. ------------
        // D layout: lane = n%16 + 16*m_hi, vgpr r holds M = r + 8*m_hi.
#pragma unroll
        for (int r = 0; r < 8; ++r) {
            int m = m0 + r + 8 * hf;
            float2 xi   = *(const float2*)(xin + m * NINP);
            float drive = xi.x * w0 + xi.y * w1 + bn
                        + nz[(size_t)m * HID + n] * SIGMA_REC;
            float sp    = prev[(size_t)m * HID + n];
            next[(size_t)m * HID + n] = OMA * sp + ALPHA * (c[r] + drive);
            // Prefetch next step's noise stream (global_prefetch_b8);
            // speculative — dropped harmlessly if it misses translation.
            if (t + 1 < T_STEPS)
                __builtin_prefetch(nz + (size_t)BATCH * HID + (size_t)m * HID + n, 0, 1);
        }

        // ---- Device-scope grid barrier between steps. --------------------
        // Release: publish slab t+1 writes (global_wb). Acquire: invalidate
        // WGP-local cache before re-reading slab t+1 next iteration
        // (global_inv). Counter is monotonic: target = NBLOCKS * (t+1).
        __syncthreads();   // also protects As reuse (all WMMA reads done)
        if (tid == 0) {
            __hip_atomic_fetch_add(bar_cnt, 1u, __ATOMIC_RELEASE,
                                   __HIP_MEMORY_SCOPE_AGENT);
            const unsigned target = (unsigned)NBLOCKS * (unsigned)(t + 1);
            while (__hip_atomic_load(bar_cnt, __ATOMIC_ACQUIRE,
                                     __HIP_MEMORY_SCOPE_AGENT) < target) {
                __builtin_amdgcn_s_sleep(2);
            }
        }
        __syncthreads();
    }
}

// ---------------------------------------------------------------------------
extern "C" void kernel_launch(void* const* d_in, const int* in_sizes, int n_in,
                              void* d_out, int out_size, void* d_ws, size_t ws_size,
                              hipStream_t stream) {
    const float* inputs = (const float*)d_in[0];  // [T][B][2]
    const float* init   = (const float*)d_in[1];  // [B][H]
    const float* noise  = (const float*)d_in[2];  // [T][B][H]
    const float* wih    = (const float*)d_in[3];  // [2][H]
    const float* whh    = (const float*)d_in[4];  // [H][H]
    const float* bias   = (const float*)d_in[5];  // [H]
    float* out = (float*)d_out;                   // [T+1][B][H]

    // Workspace layout: [0,256) barrier counter (+pad), [256, 256+2MB) Wb.
    unsigned* bar_cnt = (unsigned*)d_ws;
    bf16*     Wb      = (bf16*)((char*)d_ws + 256);

    // Reset barrier counter every call (graph-capture-safe memset node).
    hipMemsetAsync(bar_cnt, 0, 256, stream);

    // One-time weight swizzle (deterministic; ~6 MB of traffic, negligible).
    swizzle_whh<<<(NT * KT * 32) / 256, 256, 0, stream>>>(whh, Wb);

    // Output slab 0 = initial state (recurrence source for step 0).
    hipMemcpyAsync(out, init, (size_t)BATCH * HID * sizeof(float),
                   hipMemcpyDeviceToDevice, stream);

    // One persistent kernel runs all 512 steps with internal grid barriers.
    rnn_persistent<<<NBLOCKS, 256, 0, stream>>>(
        inputs, out, Wb, noise, wih, bias, bar_cnt);
}
